// GATModelBen_27152783245345
// MI455X (gfx1250) — matrix operand order repaired and checked by reading, not compiled, not run on hardware
//
#include <hip/hip_runtime.h>
#include <hip/hip_bf16.h>
#include <math.h>

// GAT (2x GATConv + conv1d head + log_softmax) for MI455X / gfx1250.
// GEMMs: LDS-staged B (f16, transposed) + v_wmma_f32_16x16x32_f16.
// Edge softmax/scatter via L2-resident float atomics (h/agg fit in 192MB L2).

#define DIN    256
#define HEADS  8
#define CH     32
#define HID    256
#define NOUT   40
#define NEG_SLOPE 0.2f

#define GEMM_K   256          // K of every GEMM in this model (DIN == HID)
#define LDSK     264          // padded column stride: 528B -> bank stride 4
#define BLK_M    128          // rows per block (8 waves x 16)
#define BLK_N    64           // cols per block (4 WMMA sub-tiles)

typedef __attribute__((ext_vector_type(16))) _Float16 v16h;
typedef __attribute__((ext_vector_type(8)))  _Float16 v8h;
typedef __attribute__((ext_vector_type(8)))  float    v8f;

// ---------------------------------------------------------------------------
// WMMA GEMM: D[M,Nn] = A[M,K=256] x B[256,Nn] (+ optional bias).
// Block = 256 threads = 8 waves; block tile = 128x64. B tile is staged once
// into LDS as f16, transposed to Bs[col][k] (stride LDSK), so each lane's
// B fragment is two contiguous 16B ds_load_b128 reads. A is read from global
// in contiguous 32B chunks (clauses of global_load_b128). Out-of-range rows
// and columns are clamped on load and masked on store (no divergent loads).
// ---------------------------------------------------------------------------
__global__ void __launch_bounds__(256)
wmma_gemm_kernel(const float* __restrict__ A,
                 const float* __restrict__ B,
                 const float* __restrict__ bias,
                 float* __restrict__ D,
                 int M, int Nn) {
  __shared__ _Float16 Bs[BLK_N * LDSK];     // ~33 KB

  const int tid  = threadIdx.x;
  const int lane = tid & 31;
  const int wave = tid >> 5;
  const int ntn  = (Nn + BLK_N - 1) / BLK_N;
  const int bm   = blockIdx.x / ntn;
  const int bn   = blockIdx.x - bm * ntn;

  // ---- cooperative B stage: global f32 -> LDS f16 transposed -------------
  // flat = k*64 + c; consecutive tid -> consecutive columns (coalesced).
  for (int flat = tid; flat < GEMM_K * BLK_N; flat += 256) {
    const int k = flat >> 6;
    const int c = flat & 63;
    const int gc  = bn * BLK_N + c;
    const int gcc = gc < Nn ? gc : (Nn - 1);          // clamp, masked later
    Bs[c * LDSK + k] = (_Float16)B[k * Nn + gcc];
  }
  __syncthreads();

  const int half = lane >> 4;               // 0: lanes 0-15, 1: lanes 16-31
  const int l16  = lane & 15;
  const int arow0 = bm * BLK_M + wave * 16 + l16;    // A row for this lane
  const int arow  = arow0 < M ? arow0 : (M - 1);     // clamp (masked store)
  const float* __restrict__ Ar = A + arow * GEMM_K;

  v8f acc[4] = {};
#pragma unroll 4
  for (int kk = 0; kk < GEMM_K; kk += 32) {
    // A fragment (16x32 f16): element i -> K = kk + 16*half + i
    v16h a;
    const int ka0 = kk + half * 16;
#pragma unroll
    for (int i = 0; i < 8; ++i) {
      a[i]     = (_Float16)Ar[ka0 + i];
      a[i + 8] = (_Float16)Ar[ka0 + 8 + i];
    }
    // B fragments from LDS: element i -> K = kk + 16*half + i, col = j*16+l16
    const int kb0 = kk + half * 16;
#pragma unroll
    for (int j = 0; j < 4; ++j) {
      const _Float16* __restrict__ Bc = &Bs[(j * 16 + l16) * LDSK + kb0];
      const v8h b0 = *(const v8h*)(Bc);     // 16B aligned ds_load_b128
      const v8h b1 = *(const v8h*)(Bc + 8); // 16B aligned ds_load_b128
      v16h b;
#pragma unroll
      for (int i = 0; i < 8; ++i) { b[i] = b0[i]; b[i + 8] = b1[i]; }
      acc[j] = __builtin_amdgcn_wmma_f32_16x16x32_f16(
          /*neg_a=*/false, a, /*neg_b=*/false, b,
          /*c_mod=*/(short)0, acc[j], /*reuse_a=*/false, /*reuse_b=*/false);
    }
  }

  const int r0 = bm * BLK_M + wave * 16 + half * 8;  // D: VGPR v -> M = r0+v
#pragma unroll
  for (int j = 0; j < 4; ++j) {
    const int ncol = bn * BLK_N + j * 16 + l16;
    if (ncol < Nn) {
      const float bv = (bias != nullptr) ? bias[ncol] : 0.0f;
#pragma unroll
      for (int v = 0; v < 8; ++v) {
        const int r = r0 + v;
        if (r < M)
          D[r * Nn + ncol] = acc[j][v] + bv;
      }
    }
  }
}

// ---------------------------------------------------------------------------
// Per-node attention logits: asrc[n,h] = <h[n,h,:], a_s[h,:]>, same for adst.
// ---------------------------------------------------------------------------
__global__ void attn_node_kernel(const float* __restrict__ Hf,
                                 const float* __restrict__ a_s,
                                 const float* __restrict__ a_d,
                                 float* __restrict__ asrc,
                                 float* __restrict__ adst, int n) {
  const int idx = blockIdx.x * blockDim.x + threadIdx.x;
  if (idx >= n * HEADS) return;
  const int node = idx >> 3, h = idx & 7;
  const float* __restrict__ hp  = Hf + node * HID + h * CH;
  const float* __restrict__ asp = a_s + h * CH;
  const float* __restrict__ adp = a_d + h * CH;
  float vs = 0.f, vd = 0.f;
#pragma unroll
  for (int c = 0; c < CH; ++c) { const float x = hp[c]; vs += x * asp[c]; vd += x * adp[c]; }
  asrc[idx] = vs;
  adst[idx] = vd;
}

// ---------------------------------------------------------------------------
// Per-layer init: agg = 0, amax = -inf, denom = 0 (capture-safe, no memset).
// ---------------------------------------------------------------------------
__global__ void init_layer_kernel(float* __restrict__ agg,
                                  float* __restrict__ amax,
                                  float* __restrict__ denom, int n) {
  const int i = blockIdx.x * blockDim.x + threadIdx.x;
  if (i < n * HID)   agg[i] = 0.f;
  if (i < n * HEADS) { amax[i] = -INFINITY; denom[i] = 0.f; }
}

__device__ __forceinline__ void edge_ends(const long long* __restrict__ ei,
                                          int E, int e, int& s, int& d) {
  if (e < E) { s = (int)ei[e]; d = (int)ei[(size_t)E + e]; }
  else       { s = e - E; d = e - E; }          // appended self loops
}

__device__ __forceinline__ float leaky(float a) {
  return a > 0.f ? a : NEG_SLOPE * a;
}

// signbit-split float atomic max: exact for every interleaving, init = -inf.
__device__ __forceinline__ void atomic_max_f32(float* addr, float val) {
  if (!__builtin_signbitf(val))
    atomicMax((int*)addr, __float_as_int(val));
  else
    atomicMin((unsigned int*)addr, __float_as_uint(val));
}

// ---------------------------------------------------------------------------
// Edge pass 1: segment max of leaky_relu(asrc[src]+adst[dst]) over dst.
// ---------------------------------------------------------------------------
__global__ void edge_max_kernel(const long long* __restrict__ ei,
                                const float* __restrict__ asrc,
                                const float* __restrict__ adst,
                                float* __restrict__ amax, int E, int Etot) {
  const int t = blockIdx.x * blockDim.x + threadIdx.x;
  if (t >= Etot * HEADS) return;
  const int e = t >> 3, h = t & 7;
  int s, d; edge_ends(ei, E, e, s, d);
  const float a = leaky(asrc[s * HEADS + h] + adst[d * HEADS + h]);
  atomic_max_f32(&amax[d * HEADS + h], a);
}

// ---------------------------------------------------------------------------
// Edge pass 2: denom[dst,h] += exp(alpha - amax[dst,h]).
// ---------------------------------------------------------------------------
__global__ void edge_sum_kernel(const long long* __restrict__ ei,
                                const float* __restrict__ asrc,
                                const float* __restrict__ adst,
                                const float* __restrict__ amax,
                                float* __restrict__ denom, int E, int Etot) {
  const int t = blockIdx.x * blockDim.x + threadIdx.x;
  if (t >= Etot * HEADS) return;
  const int e = t >> 3, h = t & 7;
  int s, d; edge_ends(ei, E, e, s, d);
  const float a  = leaky(asrc[s * HEADS + h] + adst[d * HEADS + h]);
  const float ev = __expf(a - amax[d * HEADS + h]);
  atomicAdd(&denom[d * HEADS + h], ev);
}

// ---------------------------------------------------------------------------
// Edge pass 3: wave per edge. Lane l computes coef of head (l&7); shfl
// broadcasts; each lane streams 8 coalesced channels into agg[dst] atomics.
// global_prefetch pulls the h[src]/agg[dst] rows toward cache early.
// ---------------------------------------------------------------------------
__global__ void edge_scatter_kernel(const long long* __restrict__ ei,
                                    const float* __restrict__ Hf,
                                    const float* __restrict__ asrc,
                                    const float* __restrict__ adst,
                                    const float* __restrict__ amax,
                                    const float* __restrict__ denom,
                                    float* __restrict__ agg, int E, int Etot) {
  const int wid  = (blockIdx.x * blockDim.x + threadIdx.x) >> 5;
  const int lane = threadIdx.x & 31;
  if (wid >= Etot) return;                   // wave-uniform
  int s, d; edge_ends(ei, E, wid, s, d);
  const float* __restrict__ hs = Hf + s * HID;
  float* __restrict__ ag = agg + d * HID;
  __builtin_prefetch(hs + lane * 8, 0, 0);   // global_prefetch_b8: h[src] row
  __builtin_prefetch(ag + lane * 8, 1, 0);   // and the destination row
  const int h = lane & 7;
  const float a    = leaky(asrc[s * HEADS + h] + adst[d * HEADS + h]);
  const float ev   = __expf(a - amax[d * HEADS + h]);
  const float coef = ev / (denom[d * HEADS + h] + 1e-16f);
#pragma unroll
  for (int k = 0; k < HEADS; ++k) {
    const float ck = __shfl(coef, k, 32);    // lane k holds head k's coef
    const int f = k * 32 + lane;             // head = k, channel = lane
    atomicAdd(&ag[f], hs[f] * ck);
  }
}

// ---------------------------------------------------------------------------
// out = relu(in + bias[col]) over [n, HID].
// ---------------------------------------------------------------------------
__global__ void bias_relu_kernel(const float* __restrict__ in,
                                 const float* __restrict__ bias,
                                 float* __restrict__ out, int n) {
  const int i = blockIdx.x * blockDim.x + threadIdx.x;
  if (i >= n * HID) return;
  const float v = in[i] + bias[i & (HID - 1)];
  out[i] = v > 0.f ? v : 0.f;
}

// ---------------------------------------------------------------------------
// Row-wise log_softmax over NOUT=40 columns; one thread per row.
// ---------------------------------------------------------------------------
__global__ void log_softmax_kernel(const float* __restrict__ logits,
                                   float* __restrict__ out, int n) {
  const int i = blockIdx.x * blockDim.x + threadIdx.x;
  if (i >= n) return;
  const float* __restrict__ r = logits + i * NOUT;
  float m = -INFINITY;
#pragma unroll
  for (int j = 0; j < NOUT; ++j) m = fmaxf(m, r[j]);
  float ssum = 0.f;
#pragma unroll
  for (int j = 0; j < NOUT; ++j) ssum += __expf(r[j] - m);
  const float lse = m + __logf(ssum);
  float* __restrict__ o = out + i * NOUT;
#pragma unroll
  for (int j = 0; j < NOUT; ++j) o[j] = r[j] - lse;
}

// ---------------------------------------------------------------------------
static inline int cdiv_i(long long a, long long b) { return (int)((a + b - 1) / b); }

static inline int gemm_grid(int M, int Nn) {
  return cdiv_i(M, BLK_M) * cdiv_i(Nn, BLK_N);
}

static void run_gat_layer(const float* xin,
                          const float* W, const float* a_s, const float* a_d,
                          const float* bias, const long long* ei,
                          float* h, float* agg, float* asrc, float* adst,
                          float* amax, float* denom, float* xout,
                          int n, int E, int Etot, hipStream_t stream) {
  wmma_gemm_kernel<<<gemm_grid(n, HID), 256, 0, stream>>>(xin, W, nullptr, h, n, HID);
  init_layer_kernel<<<cdiv_i((long long)n * HID, 256), 256, 0, stream>>>(agg, amax, denom, n);
  attn_node_kernel<<<cdiv_i((long long)n * HEADS, 256), 256, 0, stream>>>(h, a_s, a_d, asrc, adst, n);
  edge_max_kernel<<<cdiv_i((long long)Etot * HEADS, 256), 256, 0, stream>>>(ei, asrc, adst, amax, E, Etot);
  edge_sum_kernel<<<cdiv_i((long long)Etot * HEADS, 256), 256, 0, stream>>>(ei, asrc, adst, amax, denom, E, Etot);
  edge_scatter_kernel<<<cdiv_i((long long)Etot * 32, 256), 256, 0, stream>>>(ei, h, asrc, adst, amax, denom, agg, E, Etot);
  bias_relu_kernel<<<cdiv_i((long long)n * HID, 256), 256, 0, stream>>>(agg, bias, xout, n);
}

extern "C" void kernel_launch(void* const* d_in, const int* in_sizes, int n_in,
                              void* d_out, int out_size, void* d_ws, size_t ws_size,
                              hipStream_t stream) {
  const float*     x   = (const float*)d_in[0];
  const long long* ei  = (const long long*)d_in[1];   // int64 [2, E]
  const float*     W1  = (const float*)d_in[2];
  const float*     as1 = (const float*)d_in[3];
  const float*     ad1 = (const float*)d_in[4];
  const float*     b1  = (const float*)d_in[5];
  const float*     W2  = (const float*)d_in[6];
  const float*     as2 = (const float*)d_in[7];
  const float*     ad2 = (const float*)d_in[8];
  const float*     b2  = (const float*)d_in[9];
  const float*     Wc  = (const float*)d_in[10];
  const float*     bc  = (const float*)d_in[11];

  const int n    = in_sizes[0] / DIN;
  const int E    = in_sizes[1] / 2;
  const int Etot = E + n;              // self loops appended

  // Workspace carve-up (floats): ~117 MB total for N=50k.
  float* ws     = (float*)d_ws;
  float* h      = ws;  ws += (size_t)n * HID;
  float* agg    = ws;  ws += (size_t)n * HID;
  float* asrc   = ws;  ws += (size_t)n * HEADS;
  float* adst   = ws;  ws += (size_t)n * HEADS;
  float* amax   = ws;  ws += (size_t)n * HEADS;
  float* denom  = ws;  ws += (size_t)n * HEADS;
  float* logits = ws;  ws += (size_t)n * NOUT;

  // Layer 1: x[DIN] -> agg (xout written in-place over agg; GEMM consumes
  // its input before init zeroes agg, and bias_relu is elementwise-safe).
  run_gat_layer(x, W1, as1, ad1, b1, ei, h, agg, asrc, adst, amax, denom,
                agg, n, E, Etot, stream);
  // Layer 2: agg[HID] -> agg (h buffer reused for the GEMM output).
  run_gat_layer(agg, W2, as2, ad2, b2, ei, h, agg, asrc, adst, amax, denom,
                agg, n, E, Etot, stream);

  // Head: logits = x3 @ Wc + bc, then row-wise log_softmax into d_out.
  wmma_gemm_kernel<<<gemm_grid(n, NOUT), 256, 0, stream>>>(agg, Wc, bc, logits, n, NOUT);
  log_softmax_kernel<<<cdiv_i(n, 256), 256, 0, stream>>>(logits, (float*)d_out, n);
}
